// Net_11081015623785
// MI455X (gfx1250) — matrix operand order repaired and checked by reading, not compiled, not run on hardware
//
#include <hip/hip_runtime.h>
#include <hip/hip_bf16.h>

// ---------------------------------------------------------------------------
// MI455X (gfx1250) implementation.
// Convs run as implicit GEMM on v_wmma_f32_16x16x32_bf16 (wave32):
//   A = weights [16 cout x 32 cin] per 7x7 tap (prepacked to lane layout)
//   B = activations NHWC bf16, zero-padded halo (262x262), 32ch contiguous
//   D = f32 accum, 16 cout x 64 pixels per wave (4 N-tiles).
// ---------------------------------------------------------------------------

typedef __attribute__((ext_vector_type(16))) __bf16 v16bf;
typedef __attribute__((ext_vector_type(8)))  float  v8f;

#define NPIX   131072          // B*H*W = 2*256*256
#define PW     262             // padded width/height
#define RSTRIDE 48             // raw buffer channel stride (covers 36 padded to 48)
#define EPS    1e-5f
#define TILE_ELEMS 25088       // 49 taps * 32 lanes * 16 bf16 per A tile

// ---------------- zero fill ----------------
__global__ void zero_kernel(float4* __restrict__ dst, int n) {
    int stride = gridDim.x * blockDim.x;
    for (int i = blockIdx.x * blockDim.x + threadIdx.x; i < n; i += stride)
        dst[i] = make_float4(0.f, 0.f, 0.f, 0.f);
}

// ---------------- weight prepack: f32 OIHW -> bf16 WMMA A-fragment order ----
// Layout: wpk[tile][tap][lane][16], tile = 2*L + m for res layers, 32+m for w36.
// Lane's 16 halfs: K=(lane<16?0:16)+j
__global__ void prepack_kernel(const float* __restrict__ w1,
                               const float* __restrict__ w2,
                               const float* __restrict__ w36,
                               __hip_bfloat16* __restrict__ wpk) {
    int e = blockIdx.x * blockDim.x + threadIdx.x;
    if (e >= 35 * TILE_ELEMS) return;
    int tt = e / TILE_ELEMS, r = e % TILE_ELEMS;
    int t = r / 512, r2 = r % 512;
    int lane = r2 / 16, j = r2 % 16;
    int L, m;
    if (tt < 32) { L = tt >> 1; m = tt & 1; } else { L = 16; m = tt - 32; }
    int co   = m * 16 + (lane & 15);
    int base = (lane < 16) ? 0 : 16;
    int ci   = base + j;
    float val;
    if (L < 16) {
        int i = L >> 1;
        const float* w = (L & 1) ? w2 : w1;           // even: res_w1, odd: res_w2
        val = w[(((size_t)i * 32 + co) * 32 + ci) * 49 + t];
    } else {
        val = (co < 36) ? w36[((size_t)co * 32 + ci) * 49 + t] : 0.f;
    }
    wpk[e] = __float2bfloat16(val);
}

// ---------------- flat stats (sum, sumsq) for 1-channel tensors -------------
__global__ void flat_stats_kernel(const float* __restrict__ v, int n,
                                  float* __restrict__ st) {
    __shared__ float ss[2];
    if (threadIdx.x == 0) { ss[0] = 0.f; ss[1] = 0.f; }
    __syncthreads();
    float s = 0.f, q = 0.f;
    int stride = gridDim.x * blockDim.x;
    for (int i = blockIdx.x * blockDim.x + threadIdx.x; i < n; i += stride) {
        float x = v[i]; s += x; q += x * x;
    }
    atomicAdd(&ss[0], s); atomicAdd(&ss[1], q);
    __syncthreads();
    if (threadIdx.x == 0) { atomicAdd(&st[0], ss[0]); atomicAdd(&st[48], ss[1]); }
}

// ---------------- per-channel stats of raw conv output ----------------------
#define CS_STEP 1366
__global__ void chan_stats_kernel(const float* __restrict__ raw, int nch,
                                  float* __restrict__ st) {
    __shared__ float ss[96];
    int tid = threadIdx.x;
    if (tid < 96) ss[tid] = 0.f;
    __syncthreads();
    int t = blockIdx.x * blockDim.x + tid;
    int c = t % 48, pg = t / 48;
    float s = 0.f, q = 0.f;
    if (c < nch && pg < CS_STEP) {
        for (int p = pg; p < NPIX; p += CS_STEP) {
            float x = raw[(size_t)p * RSTRIDE + c];
            s += x; q += x * x;
        }
        atomicAdd(&ss[c], s); atomicAdd(&ss[48 + c], q);
    }
    __syncthreads();
    if (tid < 96) atomicAdd(&st[tid], ss[tid]);
}

// ---------------- direct conv 1->32 with inline BN0 -------------------------
__global__ void conv1_kernel(const float* __restrict__ x,
                             const float* __restrict__ st0,
                             const float* __restrict__ g0,
                             const float* __restrict__ b0,
                             const float* __restrict__ w32,
                             float* __restrict__ raw) {
    int e = blockIdx.x * blockDim.x + threadIdx.x;
    if (e >= NPIX * 32) return;
    int c = e & 31, p = e >> 5;
    int bi = p >> 16, rem = p & 65535, y = rem >> 8, xj = rem & 255;
    float m   = st0[0] * (1.f / (float)NPIX);
    float var = st0[48] * (1.f / (float)NPIX) - m * m;
    float inv = rsqrtf(var + EPS);
    float gg  = g0[0] * inv;
    float bc  = b0[0] - m * gg;
    const float* xb = x + (size_t)bi * 65536;
    const float* wc = w32 + c * 49;
    float s = 0.f;
    for (int ky = 0; ky < 7; ++ky) {
        int yy = y + ky - 3;
        if (yy < 0 || yy > 255) continue;
        for (int kx = 0; kx < 7; ++kx) {
            int xx = xj + kx - 3;
            if (xx < 0 || xx > 255) continue;
            float v = xb[yy * 256 + xx];
            s += (gg * v + bc) * wc[ky * 7 + kx];
        }
    }
    raw[(size_t)p * RSTRIDE + c] = s;
}

// ---------------- WMMA implicit-GEMM 7x7 conv, Cin=32 -----------------------
// grid.x = B*H (one image row per block). blockDim = 128*nTilesM? no:
//   blockDim = 256 (nTilesM=2) or 384 (nTilesM=3); wave w: m = w % nTilesM,
//   xgroup = w / nTilesM covers 64 pixels.
__global__ void convw_kernel(const __hip_bfloat16* __restrict__ act,
                             const __hip_bfloat16* __restrict__ wpkL,
                             float* __restrict__ raw, int nTilesM) {
    const int tid  = threadIdx.x;
    const int wave = tid >> 5;
    const int lane = tid & 31;
    const int l16  = lane & 15;
    const int hi   = lane >> 4;            // channel group / M half
    const int m    = wave % nTilesM;       // cout tile
    const int xg   = wave / nTilesM;       // 0..3
    const int bi   = blockIdx.x >> 8;
    const int y0   = blockIdx.x & 255;
    const int xbase = xg * 64;

    const v16bf* wp = (const v16bf*)(wpkL + (size_t)m * TILE_ELEMS) + lane;
    const __hip_bfloat16* abase =
        act + ((size_t)(bi * PW + y0) * PW + xbase + l16) * 32 + hi * 16;

    v8f acc0 = {}, acc1 = {}, acc2 = {}, acc3 = {};

    for (int ky = 0; ky < 7; ++ky) {
        const __hip_bfloat16* rowp = abase + (size_t)ky * PW * 32;
        for (int kx = 0; kx < 7; ++kx) {
            int t = ky * 7 + kx;
            v16bf a  = wp[t * 32];
            const __hip_bfloat16* bp = rowp + kx * 32;
            v16bf b0 = *(const v16bf*)(bp);
            v16bf b1 = *(const v16bf*)(bp + 16 * 32);
            v16bf b2 = *(const v16bf*)(bp + 32 * 32);
            v16bf b3 = *(const v16bf*)(bp + 48 * 32);
            acc0 = __builtin_amdgcn_wmma_f32_16x16x32_bf16(false, a, false, b0,
                                                           (short)0, acc0, false, false);
            acc1 = __builtin_amdgcn_wmma_f32_16x16x32_bf16(false, a, false, b1,
                                                           (short)0, acc1, false, false);
            acc2 = __builtin_amdgcn_wmma_f32_16x16x32_bf16(false, a, false, b2,
                                                           (short)0, acc2, false, false);
            acc3 = __builtin_amdgcn_wmma_f32_16x16x32_bf16(false, a, false, b3,
                                                           (short)0, acc3, false, false);
        }
    }

    // D layout: lanes 0-15 -> M=r, lanes 16-31 -> M=8+r, N = lane%16
    int pbase = (bi << 16) + (y0 << 8) + xbase + l16;
    int coutBase = m * 16 + hi * 8;
    float* r0 = raw + (size_t)pbase * RSTRIDE + coutBase;
#pragma unroll
    for (int r = 0; r < 8; ++r) {
        r0[(size_t)(0 * 16) * RSTRIDE + r] = acc0[r];
        r0[(size_t)(1 * 16) * RSTRIDE + r] = acc1[r];
        r0[(size_t)(2 * 16) * RSTRIDE + r] = acc2[r];
        r0[(size_t)(3 * 16) * RSTRIDE + r] = acc3[r];
    }
}

// ---------------- BN + ReLU (+residual) -> padded bf16 NHWC -----------------
__global__ void bn_apply_kernel(const float* __restrict__ raw,
                                const float* __restrict__ st,
                                const float* __restrict__ g,
                                const float* __restrict__ bta,
                                const __hip_bfloat16* __restrict__ resid,
                                __hip_bfloat16* __restrict__ out, int nch) {
    int e = blockIdx.x * blockDim.x + threadIdx.x;
    if (e >= NPIX * nch) return;
    int c = e % nch, p = e / nch;
    float m   = st[c] * (1.f / (float)NPIX);
    float var = st[48 + c] * (1.f / (float)NPIX) - m * m;
    float inv = rsqrtf(var + EPS);
    float val = g[c] * (raw[(size_t)p * RSTRIDE + c] - m) * inv + bta[c];
    int bi = p >> 16, rem = p & 65535, y = rem >> 8, xj = rem & 255;
    size_t pidx = ((size_t)(bi * PW + y + 3) * PW + (xj + 3)) * 32 + c;
    if (resid) val += __bfloat162float(resid[pidx]);
    val = fmaxf(val, 0.f);
    out[pidx] = __float2bfloat16(val);
}

// ---------------- BN36 + ReLU + channel-mean + stats of u -------------------
__global__ void bn36_mean_kernel(const float* __restrict__ raw,
                                 const float* __restrict__ st,
                                 const float* __restrict__ g,
                                 const float* __restrict__ bta,
                                 float* __restrict__ u,
                                 float* __restrict__ ust) {
    __shared__ float ss[2];
    if (threadIdx.x == 0) { ss[0] = 0.f; ss[1] = 0.f; }
    __syncthreads();
    int p = blockIdx.x * blockDim.x + threadIdx.x;
    float uu = 0.f;
    if (p < NPIX) {
        float s = 0.f;
        for (int c = 0; c < 36; ++c) {
            float m   = st[c] * (1.f / (float)NPIX);
            float var = st[48 + c] * (1.f / (float)NPIX) - m * m;
            float inv = rsqrtf(var + EPS);
            float val = g[c] * (raw[(size_t)p * RSTRIDE + c] - m) * inv + bta[c];
            s += fmaxf(val, 0.f);
        }
        uu = s * (1.f / 36.f);
        u[p] = uu;
    }
    atomicAdd(&ss[0], uu); atomicAdd(&ss[1], uu * uu);
    __syncthreads();
    if (threadIdx.x == 0) { atomicAdd(&ust[0], ss[0]); atomicAdd(&ust[48], ss[1]); }
}

// ---------------- final: BN(u) + x, circular mask ---------------------------
__global__ void final_kernel(const float* __restrict__ u,
                             const float* __restrict__ ust,
                             const float* __restrict__ gu,
                             const float* __restrict__ bu,
                             const float* __restrict__ x,
                             float* __restrict__ out) {
    int p = blockIdx.x * blockDim.x + threadIdx.x;
    if (p >= NPIX) return;
    float m   = ust[0] * (1.f / (float)NPIX);
    float var = ust[48] * (1.f / (float)NPIX) - m * m;
    float inv = rsqrtf(var + EPS);
    float val = gu[0] * (u[p] - m) * inv + bu[0] + x[p];
    int rem = p & 65535;
    float fy = (float)(rem >> 8) - 128.f;
    float fx = (float)(rem & 255) - 128.f;
    out[p] = (fy * fy + fx * fx <= 16385.f) ? val : 0.f;
}

// ---------------------------------------------------------------------------
extern "C" void kernel_launch(void* const* d_in, const int* in_sizes, int n_in,
                              void* d_out, int out_size, void* d_ws, size_t ws_size,
                              hipStream_t stream) {
    const float* x   = (const float*)d_in[0];
    const float* g0  = (const float*)d_in[1];
    const float* b0  = (const float*)d_in[2];
    const float* w32 = (const float*)d_in[3];
    const float* g32 = (const float*)d_in[4];
    const float* b32 = (const float*)d_in[5];
    const float* rw1 = (const float*)d_in[6];
    const float* rg1 = (const float*)d_in[7];
    const float* rb1 = (const float*)d_in[8];
    const float* rw2 = (const float*)d_in[9];
    const float* rg2 = (const float*)d_in[10];
    const float* rb2 = (const float*)d_in[11];
    const float* w36 = (const float*)d_in[12];
    const float* g36 = (const float*)d_in[13];
    const float* b36 = (const float*)d_in[14];
    const float* gu  = (const float*)d_in[15];
    const float* bu  = (const float*)d_in[16];
    float* out = (float*)d_out;

    const size_t STATS_BYTES = 8192;
    const size_t PAD_BYTES   = (size_t)2 * PW * PW * 32 * 2;     // 8,786,432
    const size_t RAW_BYTES   = (size_t)NPIX * RSTRIDE * 4;       // 25,165,824
    const size_t U_BYTES     = (size_t)NPIX * 4;

    char* ws = (char*)d_ws;
    float*          stats = (float*)ws;
    __hip_bfloat16* padA  = (__hip_bfloat16*)(ws + STATS_BYTES);
    __hip_bfloat16* padB  = (__hip_bfloat16*)(ws + STATS_BYTES + PAD_BYTES);
    float*          raw   = (float*)(ws + STATS_BYTES + 2 * PAD_BYTES);
    float*          ubuf  = (float*)(ws + STATS_BYTES + 2 * PAD_BYTES + RAW_BYTES);
    __hip_bfloat16* wpk   = (__hip_bfloat16*)(ws + STATS_BYTES + 2 * PAD_BYTES +
                                              RAW_BYTES + U_BYTES);

    // 1) zero halo buffers + stats (atomic targets)
    {
        int n16 = (int)((2 * PAD_BYTES) / 16);
        zero_kernel<<<2048, 256, 0, stream>>>((float4*)padA, n16);
        zero_kernel<<<8, 256, 0, stream>>>((float4*)stats, (int)(STATS_BYTES / 16));
    }
    // 2) prepack all WMMA weights to bf16 A-fragment order
    {
        int total = 35 * TILE_ELEMS;
        prepack_kernel<<<(total + 255) / 256, 256, 0, stream>>>(rw1, rw2, w36, wpk);
    }
    // 3) BN0 stats over x
    flat_stats_kernel<<<256, 256, 0, stream>>>(x, NPIX, stats + 0 * 96);
    // 4) conv 1->32 (direct, BN0 fused)
    conv1_kernel<<<(NPIX * 32 + 255) / 256, 256, 0, stream>>>(x, stats, g0, b0, w32, raw);
    chan_stats_kernel<<<257, 256, 0, stream>>>(raw, 32, stats + 1 * 96);
    bn_apply_kernel<<<(NPIX * 32 + 255) / 256, 256, 0, stream>>>(
        raw, stats + 1 * 96, g32, b32, nullptr, padA, 32);

    // 5) 8 residual blocks, 2 WMMA convs each
    for (int i = 0; i < 8; ++i) {
        int L1 = 2 * i, L2 = 2 * i + 1;
        convw_kernel<<<512, 256, 0, stream>>>(padA, wpk + (size_t)L1 * 2 * TILE_ELEMS, raw, 2);
        chan_stats_kernel<<<257, 256, 0, stream>>>(raw, 32, stats + (2 + L1) * 96);
        bn_apply_kernel<<<(NPIX * 32 + 255) / 256, 256, 0, stream>>>(
            raw, stats + (2 + L1) * 96, rg1 + i * 32, rb1 + i * 32, nullptr, padB, 32);

        convw_kernel<<<512, 256, 0, stream>>>(padB, wpk + (size_t)L2 * 2 * TILE_ELEMS, raw, 2);
        chan_stats_kernel<<<257, 256, 0, stream>>>(raw, 32, stats + (2 + L2) * 96);
        bn_apply_kernel<<<(NPIX * 32 + 255) / 256, 256, 0, stream>>>(
            raw, stats + (2 + L2) * 96, rg2 + i * 32, rb2 + i * 32, padA, padA, 32);
    }

    // 6) conv 32->36 (3 M-tiles, rows 36..47 zero-weighted)
    convw_kernel<<<512, 384, 0, stream>>>(padA, wpk + (size_t)32 * TILE_ELEMS, raw, 3);
    chan_stats_kernel<<<257, 256, 0, stream>>>(raw, 36, stats + 18 * 96);
    // 7) BN36 + ReLU + channel-mean + u stats
    bn36_mean_kernel<<<512, 256, 0, stream>>>(raw, stats + 18 * 96, g36, b36,
                                              ubuf, stats + 19 * 96);
    // 8) BN(u) + x + circular mask
    final_kernel<<<512, 256, 0, stream>>>(ubuf, stats + 19 * 96, gu, bu, x, out);
}
